// BasicLSTM_43894565765582
// MI455X (gfx1250) — compile-verified, matrix operand order checked by reading, not run on hardware
//
#include <hip/hip_runtime.h>

// ---------------------------------------------------------------------------
// Fused LSTM step for MI455X (gfx1250, wave32), two stages:
//   Stage 1: pack fp32 -> bf16 (inputs, h0, 6 weight mats) into d_ws.
//            Working set (~28 MB bf16) then lives in the 192 MB L2.
//   Stage 2: fused 3-gate GEMM on v_wmma_f32_16x16x32_bf16, A-tiles staged
//            via async global->LDS (ASYNCcnt) when the builtin is available,
//            fused sigmoid/tanh cell update epilogue.
// Fallback (tiny d_ws): single fused fp32-input kernel (cvt in-loop).
// ---------------------------------------------------------------------------

typedef __attribute__((ext_vector_type(16))) __bf16   bf16x16;
typedef __attribute__((ext_vector_type(8)))  __bf16   bf16x8;
typedef __attribute__((ext_vector_type(4)))  __bf16   bf16x4;
typedef __attribute__((ext_vector_type(8)))  float    f32x8;
typedef __attribute__((ext_vector_type(4)))  float    f32x4;
typedef __attribute__((ext_vector_type(4)))  unsigned uint32x4;
typedef __attribute__((ext_vector_type(4)))  int      v4i;

// Address-space-qualified int4 pointers for the async global->LDS builtin:
// param 0 is AS(1) ("__device__") v4i*, param 1 is AS(3) ("__shared__") v4i*.
typedef __attribute__((address_space(1))) v4i* gptr_v4i;
typedef __attribute__((address_space(3))) v4i* lptr_v4i;

#define BM 128          // block tile rows (batch)
#define BN 64           // block tile cols (state)
#define BK 32           // K per WMMA step (bf16 WMMA K)
#define D_DIM 1024
#define S_DIM 1024
#define NTHREADS 256    // 8 waves of 32

#if defined(__AMDGCN__) &&                                         \
    __has_builtin(__builtin_amdgcn_global_load_async_to_lds_b128) && \
    __has_builtin(__builtin_amdgcn_s_wait_asynccnt)
#define USE_ASYNC_COPY 1
#else
#define USE_ASYNC_COPY 0
#endif

struct SharedTiles {
    __bf16 A[BM][BK];         // 128 x 32 bf16  = 8 KB   (row-major, k contiguous)
    __bf16 Bt[3][BN][BK];     // 3 x 64 x 32    = 12 KB  ([gate][n][k], k contiguous)
};

// Assemble a bf16 WMMA fragment from an LDS row of 32 bf16.
// ISA 16-bit A/B layout: lanes 0-15 hold K {0..7,16..23}, lanes 16-31 hold
// K {8..15,24..31}; VGPR v packs K pair (2v,2v+1) within its group.
__device__ __forceinline__ bf16x16 load_frag(const __bf16* rowp, int half) {
    union { bf16x16 v; unsigned u[8]; } f;
    const unsigned* p = (const unsigned*)rowp;   // 16 dwords per 32-elem row
    const int o1 = half ? 4 : 0;
    const int o2 = half ? 12 : 8;
#pragma unroll
    for (int v = 0; v < 4; ++v) f.u[v] = p[o1 + v];
#pragma unroll
    for (int v = 0; v < 4; ++v) f.u[4 + v] = p[o2 + v];
    return f.v;
}

__device__ __forceinline__ float sigmoid_fast(float x) {
    return 1.0f / (1.0f + __expf(-x));
}
__device__ __forceinline__ float tanh_fast(float x) {
    return 1.0f - 2.0f / (__expf(2.0f * x) + 1.0f);
}

// 2x2 WMMA tiles per wave, 3 gates, for the current LDS tile.
__device__ __forceinline__ void wmma_step(SharedTiles& sm, f32x8 (&acc)[3][2][2]) {
    const int lane  = threadIdx.x & 31;
    const int wid   = threadIdx.x >> 5;
    const int waveM = wid >> 1;
    const int waveN = wid & 1;
    const int half  = lane >> 4;
    const int l15   = lane & 15;

    bf16x16 a0 = load_frag(&sm.A[waveM * 32 + l15][0], half);
    bf16x16 a1 = load_frag(&sm.A[waveM * 32 + 16 + l15][0], half);
#pragma unroll
    for (int g = 0; g < 3; ++g) {
        bf16x16 b0 = load_frag(&sm.Bt[g][waveN * 32 + l15][0], half);
        bf16x16 b1 = load_frag(&sm.Bt[g][waveN * 32 + 16 + l15][0], half);
        acc[g][0][0] = __builtin_amdgcn_wmma_f32_16x16x32_bf16(
            false, a0, false, b0, (short)0, acc[g][0][0], false, false);
        acc[g][0][1] = __builtin_amdgcn_wmma_f32_16x16x32_bf16(
            false, a0, false, b1, (short)0, acc[g][0][1], false, false);
        acc[g][1][0] = __builtin_amdgcn_wmma_f32_16x16x32_bf16(
            false, a1, false, b0, (short)0, acc[g][1][0], false, false);
        acc[g][1][1] = __builtin_amdgcn_wmma_f32_16x16x32_bf16(
            false, a1, false, b1, (short)0, acc[g][1][1], false, false);
    }
}

// Fused epilogue: bias + activations + cell/hidden update.
__device__ __forceinline__ void lstm_epilogue(
    const f32x8 (&acc)[3][2][2],
    const float* __restrict__ b_f, const float* __restrict__ b_i,
    const float* __restrict__ b_o, const float* __restrict__ c0,
    float* __restrict__ out_h, float* __restrict__ out_c,
    int rowBase, int colBase)
{
    const int lane  = threadIdx.x & 31;
    const int wid   = threadIdx.x >> 5;
    const int waveM = wid >> 1;
    const int waveN = wid & 1;
    const int half  = lane >> 4;
    const int l15   = lane & 15;

#pragma unroll
    for (int mt = 0; mt < 2; ++mt) {
#pragma unroll
        for (int nt = 0; nt < 2; ++nt) {
            const int col = colBase + waveN * 32 + nt * 16 + l15;
            const float bf = b_f[col];
            const float bi = b_i[col];
            const float bo = b_o[col];
#pragma unroll
            for (int r = 0; r < 8; ++r) {
                // C/D layout: lanes 0-15 -> M=r, lanes 16-31 -> M=r+8
                const int row = rowBase + waveM * 32 + mt * 16 + half * 8 + r;
                const size_t o = (size_t)row * S_DIM + col;
                const float sf = sigmoid_fast(acc[0][mt][nt][r] + bf);
                const float si = sigmoid_fast(acc[1][mt][nt][r] + bi);
                const float so = sigmoid_fast(acc[2][mt][nt][r] + bo);
                // input_cell reuses the forget pre-activation (reference quirk)
                const float cn = sf * c0[o] + sf * si;
                const float hn = so * tanh_fast(cn);
                out_h[o] = hn;
                out_c[o] = cn;
            }
        }
    }
}

// ------------------------- Stage 1: fp32 -> bf16 pack ----------------------
__global__ __launch_bounds__(NTHREADS)
void pack_bf16_kernel(const float* __restrict__ src, __bf16* __restrict__ dst,
                      int n4)
{
    int i = blockIdx.x * blockDim.x + threadIdx.x;
    if (i >= n4) return;
    f32x4 v = ((const f32x4*)src)[i];
    bf16x4 b;
    b.x = (__bf16)v.x;
    b.y = (__bf16)v.y;
    b.z = (__bf16)v.z;
    b.w = (__bf16)v.w;
    ((bf16x4*)dst)[i] = b;
}

// ------------------- Stage 2: bf16 GEMM phase (hot path) -------------------
__device__ __forceinline__ void gemm_phase_bf16(
    const __bf16* __restrict__ A, int lda,
    const __bf16* __restrict__ W0, const __bf16* __restrict__ W1,
    const __bf16* __restrict__ W2, int K, int ldw,
    int rowBase, int colBase, SharedTiles& sm, f32x8 (&acc)[3][2][2])
{
    const int tid = threadIdx.x;
    const __bf16* Ws[3] = { W0, W1, W2 };

    for (int k0 = 0; k0 < K; k0 += BK) {
        __syncthreads();   // previous compute done before LDS overwrite

        // --- A tile: 128x32 bf16 = 512 x 16B chunks, 2 per thread ---
#pragma unroll
        for (int i = 0; i < (BM * BK) / (8 * NTHREADS); ++i) {
            int idx = i * NTHREADS + tid;
            int r  = idx >> 2;             // 4 chunks per 32-elem row
            int c8 = (idx & 3) * 8;
            const __bf16* gsrc = &A[(size_t)(rowBase + r) * lda + (k0 + c8)];
            __bf16* ldst = &sm.A[r][c8];
#if USE_ASYNC_COPY
            // ASYNCcnt-tracked DMA: global_load_async_to_lds_b128
            __builtin_amdgcn_global_load_async_to_lds_b128(
                (gptr_v4i)gsrc, (lptr_v4i)ldst, 0, 0);
#else
            *(uint32x4*)ldst = *(const uint32x4*)gsrc;
#endif
        }

        // --- B tiles: 3 x (32x64), 1 x 16B chunk per thread per gate,
        //     scattered into [n][k] (transposed) for dword fragment loads ---
        {
            int k  = tid >> 3;             // 8 chunks per 64-elem row
            int n8 = (tid & 7) * 8;
#pragma unroll
            for (int g = 0; g < 3; ++g) {
                bf16x8 v = *(const bf16x8*)
                    &Ws[g][(size_t)(k0 + k) * ldw + (colBase + n8)];
#pragma unroll
                for (int j = 0; j < 8; ++j)
                    sm.Bt[g][n8 + j][k] = v[j];
            }
        }

        // Prefetch next K tiles into L2 (global_prefetch_b8)
        if (k0 + BK < K) {
            __builtin_prefetch(&A[(size_t)(rowBase + (tid & (BM - 1))) * lda
                                  + (k0 + BK)], 0, 0);
            __builtin_prefetch(&Ws[tid % 3][(size_t)(k0 + BK + (tid & 31)) * ldw
                                  + colBase], 0, 0);
        }

#if USE_ASYNC_COPY
        __builtin_amdgcn_s_wait_asynccnt(0);   // drain async global->LDS DMA
#endif
        __syncthreads();

        wmma_step(sm, acc);
    }
}

__global__ __launch_bounds__(NTHREADS)
void lstm_gemm_bf16_kernel(
    const __bf16* __restrict__ Xb, const __bf16* __restrict__ Hb,
    const __bf16* __restrict__ Wf, const __bf16* __restrict__ Wi,
    const __bf16* __restrict__ Wo, const __bf16* __restrict__ Uf,
    const __bf16* __restrict__ Ui, const __bf16* __restrict__ Uo,
    const float* __restrict__ b_f, const float* __restrict__ b_i,
    const float* __restrict__ b_o, const float* __restrict__ c0,
    float* __restrict__ out_h, float* __restrict__ out_c)
{
    __shared__ SharedTiles sm;
    const int rowBase = blockIdx.y * BM;
    const int colBase = blockIdx.x * BN;

    f32x8 acc[3][2][2] = {};
    gemm_phase_bf16(Xb, D_DIM, Wf, Wi, Wo, D_DIM, S_DIM,
                    rowBase, colBase, sm, acc);
    gemm_phase_bf16(Hb, S_DIM, Uf, Ui, Uo, S_DIM, S_DIM,
                    rowBase, colBase, sm, acc);
    lstm_epilogue(acc, b_f, b_i, b_o, c0, out_h, out_c, rowBase, colBase);
}

// -------------- Fallback: fused fp32-input kernel (cvt in-loop) ------------
__device__ __forceinline__ void gemm_phase_f32(
    const float* __restrict__ A, int lda,
    const float* __restrict__ W0, const float* __restrict__ W1,
    const float* __restrict__ W2, int K, int ldw,
    int rowBase, int colBase, SharedTiles& sm, f32x8 (&acc)[3][2][2])
{
    const int tid = threadIdx.x;
    const float* Ws[3] = { W0, W1, W2 };

    for (int k0 = 0; k0 < K; k0 += BK) {
        __syncthreads();
#pragma unroll
        for (int i = 0; i < (BM * BK) / NTHREADS; ++i) {
            int idx = i * NTHREADS + tid;
            int r = idx >> 5, c = idx & 31;
            sm.A[r][c] = (__bf16)A[(size_t)(rowBase + r) * lda + (k0 + c)];
        }
#pragma unroll
        for (int g = 0; g < 3; ++g) {
#pragma unroll
            for (int i = 0; i < (BK * BN) / NTHREADS; ++i) {
                int idx = i * NTHREADS + tid;
                int k = idx >> 6, n = idx & 63;
                sm.Bt[g][n][k] =
                    (__bf16)Ws[g][(size_t)(k0 + k) * ldw + (colBase + n)];
            }
        }
        __syncthreads();
        wmma_step(sm, acc);
    }
}

__global__ __launch_bounds__(NTHREADS)
void lstm_fused_f32_kernel(
    const float* __restrict__ inputs, const float* __restrict__ h0,
    const float* __restrict__ c0,
    const float* __restrict__ w_f, const float* __restrict__ u_f,
    const float* __restrict__ b_f,
    const float* __restrict__ w_i, const float* __restrict__ u_i,
    const float* __restrict__ b_i,
    const float* __restrict__ w_o, const float* __restrict__ u_o,
    const float* __restrict__ b_o,
    float* __restrict__ out_h, float* __restrict__ out_c)
{
    __shared__ SharedTiles sm;
    const int rowBase = blockIdx.y * BM;
    const int colBase = blockIdx.x * BN;

    f32x8 acc[3][2][2] = {};
    gemm_phase_f32(inputs, D_DIM, w_f, w_i, w_o, D_DIM, S_DIM,
                   rowBase, colBase, sm, acc);
    gemm_phase_f32(h0, S_DIM, u_f, u_i, u_o, S_DIM, S_DIM,
                   rowBase, colBase, sm, acc);
    lstm_epilogue(acc, b_f, b_i, b_o, c0, out_h, out_c, rowBase, colBase);
}

// ---------------------------------------------------------------------------
extern "C" void kernel_launch(void* const* d_in, const int* in_sizes, int n_in,
                              void* d_out, int out_size, void* d_ws,
                              size_t ws_size, hipStream_t stream)
{
    const float* inputs = (const float*)d_in[0];
    const float* h0     = (const float*)d_in[1];
    const float* c0     = (const float*)d_in[2];
    const float* w_f    = (const float*)d_in[3];
    const float* u_f    = (const float*)d_in[4];
    const float* b_f    = (const float*)d_in[5];
    const float* w_i    = (const float*)d_in[6];
    const float* u_i    = (const float*)d_in[7];
    const float* b_i    = (const float*)d_in[8];
    const float* w_o    = (const float*)d_in[9];
    const float* u_o    = (const float*)d_in[10];
    const float* b_o    = (const float*)d_in[11];

    const int Bsz = in_sizes[0] / D_DIM;            // 4096
    float* out_h = (float*)d_out;                   // [B, S]
    float* out_c = out_h + (size_t)Bsz * S_DIM;     // [B, S]

    const size_t nX = (size_t)Bsz * D_DIM;
    const size_t nH = (size_t)Bsz * S_DIM;
    const size_t nW = (size_t)D_DIM * S_DIM;
    const size_t need = 2 * (nX + nH + 6 * nW);     // bytes of bf16 scratch

    dim3 grid(S_DIM / BN, Bsz / BM);                // 16 x 32 blocks

    if (ws_size >= need) {
        __bf16* Xb = (__bf16*)d_ws;
        __bf16* Hb = Xb + nX;
        __bf16* Wf = Hb + nH;
        __bf16* Wi = Wf + nW;
        __bf16* Wo = Wi + nW;
        __bf16* Uf = Wo + nW;
        __bf16* Ui = Uf + nW;
        __bf16* Uo = Ui + nW;

        const float* srcs[8] = { inputs, h0, w_f, w_i, w_o, u_f, u_i, u_o };
        __bf16*      dsts[8] = { Xb, Hb, Wf, Wi, Wo, Uf, Ui, Uo };
        const size_t ns[8]   = { nX, nH, nW, nW, nW, nW, nW, nW };
        for (int t = 0; t < 8; ++t) {
            int n4 = (int)(ns[t] / 4);
            pack_bf16_kernel<<<(n4 + NTHREADS - 1) / NTHREADS, NTHREADS, 0,
                               stream>>>(srcs[t], dsts[t], n4);
        }
        lstm_gemm_bf16_kernel<<<grid, NTHREADS, 0, stream>>>(
            Xb, Hb, Wf, Wi, Wo, Uf, Ui, Uo,
            b_f, b_i, b_o, c0, out_h, out_c);
    } else {
        lstm_fused_f32_kernel<<<grid, NTHREADS, 0, stream>>>(
            inputs, h0, c0, w_f, u_f, b_f, w_i, u_i, b_i, w_o, u_o, b_o,
            out_h, out_c);
    }
}
